// BasicTransformerBlock_27212912788027
// MI455X (gfx1250) — matrix operand order repaired and checked by reading, not compiled, not run on hardware
//
#include <hip/hip_runtime.h>
#include <math.h>

// ---------------------------------------------------------------------------
// BasicTransformerBlock for MI455X (gfx1250, wave32, WMMA).
// All matmuls via v_wmma_f32_16x16x32_f16 (f16 in, f32 accumulate).
// B=2, N=2048, D=512, H=8, d=64, ctx=1024.
// ---------------------------------------------------------------------------

typedef _Float16 v16h __attribute__((ext_vector_type(16)));
typedef float    v8f  __attribute__((ext_vector_type(8)));

static __device__ __forceinline__ v8f wmma16(v16h a, v16h b, v8f c) {
  return __builtin_amdgcn_wmma_f32_16x16x32_f16(false, a, false, b, (short)0, c,
                                                false, false);
}

#define DMODEL 512
#define MROWS  4096   // B * N
#define NBATCH 2
#define NTOK   2048
#define NCTX   1024
#define NHEAD  8
#define HDIM   64

// ---------------- weight convert: fp32 [K,N] -> fp16 transposed [N,K] ------
__global__ void cvtT_k(const float* __restrict__ w, _Float16* __restrict__ wt,
                       int K, int N) {
  int idx = blockIdx.x * blockDim.x + threadIdx.x;
  if (idx >= N * K) return;
  int n = idx / K, k = idx - n * K;
  wt[idx] = (_Float16)w[(size_t)k * N + n];
}

// ---------------- plain fp32 -> fp16 ---------------------------------------
__global__ void cvt_k(const float* __restrict__ a, _Float16* __restrict__ o,
                      int n) {
  int i = blockIdx.x * blockDim.x + threadIdx.x;
  if (i < n) o[i] = (_Float16)a[i];
}

// ---------------- AdaLN embeddings: ss[j][b][c] = (t[b] @ Wj + bj)[c] ------
__global__ void embed_k(const float* __restrict__ t,
                        const float* __restrict__ w0, const float* __restrict__ b0,
                        const float* __restrict__ w1, const float* __restrict__ b1,
                        const float* __restrict__ w2, const float* __restrict__ b2,
                        float* __restrict__ ss) {
  int idx = blockIdx.x * blockDim.x + threadIdx.x;   // 3 * B * 1024
  if (idx >= 3 * NBATCH * 1024) return;
  int c = idx & 1023;
  int b = (idx >> 10) & 1;
  int j = idx >> 11;
  const float* w  = (j == 0) ? w0 : (j == 1) ? w1 : w2;
  const float* bb = (j == 0) ? b0 : (j == 1) ? b1 : b2;
  const float* tp = t + b * DMODEL;
  float acc = bb[c];
  for (int k = 0; k < DMODEL; ++k) acc += tp[k] * w[(size_t)k * 1024 + c];
  ss[idx] = acc;
}

// ---------------- AdaLN apply: LN(no affine) * (1+scale) + shift -> f16 ----
__global__ __launch_bounds__(256) void adaln_k(const float* __restrict__ x,
                                               const float* __restrict__ ss,
                                               int j, _Float16* __restrict__ h) {
  int wv = threadIdx.x >> 5, lane = threadIdx.x & 31;
  int row = blockIdx.x * 8 + wv;                     // MROWS rows
  const float* xp = x + (size_t)row * DMODEL;
  int b = row / NTOK;
  const float* sc = ss + ((size_t)j * NBATCH + b) * 1024;
  float xv[16], s = 0.f, s2 = 0.f;
#pragma unroll
  for (int i = 0; i < 16; ++i) {
    float v = xp[lane + i * 32];
    xv[i] = v; s += v; s2 += v * v;
  }
#pragma unroll
  for (int off = 1; off < 32; off <<= 1) {
    s  += __shfl_xor(s,  off, 32);
    s2 += __shfl_xor(s2, off, 32);
  }
  float mu = s * (1.f / DMODEL);
  float var = s2 * (1.f / DMODEL) - mu * mu;
  float rinv = rsqrtf(var + 1e-5f);
  _Float16* hp = h + (size_t)row * DMODEL;
#pragma unroll
  for (int i = 0; i < 16; ++i) {
    int c = lane + i * 32;
    float xn = (xv[i] - mu) * rinv;
    hp[c] = (_Float16)(xn * (1.f + sc[c]) + sc[512 + c]);
  }
}

// ---------------- generic WMMA GEMM: C[M,N] = A[M,K] * Bt[N,K]^T -----------
// MODE 0: f16 out row-major.  MODE 1: f16 out V-transposed [(b*N+n)*tokB+tok].
// MODE 2: f32 out = acc + bias[n] + resid[m,n].
template <int MODE>
__global__ __launch_bounds__(256) void gemm_k(
    const _Float16* __restrict__ A, const _Float16* __restrict__ Bt,
    const float* __restrict__ bias, const float* __restrict__ resid,
    float* __restrict__ outF, _Float16* __restrict__ outH,
    int M, int N, int K, int tokB) {
  const int wv = threadIdx.x >> 5, lane = threadIdx.x & 31;
  const int gw = blockIdx.x * 8 + wv;
  const int nstrips = N >> 6;
  const int mt = gw / nstrips;
  const int ns = gw - mt * nstrips;
  if (mt >= (M >> 4)) return;
  const int row0 = mt << 4, col0 = ns << 6;
  const int mlane = lane & 15;
  const int kg = (lane >> 4) & 1;          // A: K-half group
  const int kb = kg << 4;                  // B: K base (0/16)
  v8f acc[4] = {};
  const _Float16* arow = A + (size_t)(row0 + mlane) * K;
  for (int k0 = 0; k0 < K; k0 += 32) {
    if (k0 + 64 <= K) __builtin_prefetch((const void*)(arow + k0 + 32), 0, 0);
    const _Float16* ap = arow + k0 + (kg << 4);
    v16h af;
#pragma unroll
    for (int i = 0; i < 8; ++i) { af[i] = ap[i]; af[8 + i] = ap[8 + i]; }
#pragma unroll
    for (int tI = 0; tI < 4; ++tI) {
      const _Float16* bp = Bt + (size_t)(col0 + (tI << 4) + mlane) * K + k0 + kb;
      v16h bf;
#pragma unroll
      for (int i = 0; i < 16; ++i) bf[i] = bp[i];
      acc[tI] = wmma16(af, bf, acc[tI]);
    }
  }
  const int mbase = (lane >> 4) << 3;
#pragma unroll
  for (int tI = 0; tI < 4; ++tI) {
#pragma unroll
    for (int r = 0; r < 8; ++r) {
      const int gm = row0 + mbase + r;
      const int gn = col0 + (tI << 4) + mlane;
      const float v = acc[tI][r];
      if (MODE == 0) {
        outH[(size_t)gm * N + gn] = (_Float16)v;
      } else if (MODE == 1) {
        const int bI = gm / tokB, tok = gm - bI * tokB;
        outH[((size_t)bI * N + gn) * tokB + tok] = (_Float16)v;
      } else {
        outF[(size_t)gm * N + gn] = v + bias[gn] + resid[(size_t)gm * N + gn];
      }
    }
  }
}

// ---------------- GEGLU first GEMM: u,g tiles fused, Y = u * gelu(g) -------
__global__ __launch_bounds__(256) void geglu_k(
    const _Float16* __restrict__ A, const _Float16* __restrict__ Bt, // [4096,512]
    const float* __restrict__ b1, _Float16* __restrict__ Y, int M, int K) {
  const int Nh = 2048;
  const int wv = threadIdx.x >> 5, lane = threadIdx.x & 31;
  const int gw = blockIdx.x * 8 + wv;
  const int nstrips = Nh >> 6;
  const int mt = gw / nstrips;
  const int ns = gw - mt * nstrips;
  if (mt >= (M >> 4)) return;
  const int row0 = mt << 4, col0 = ns << 6;
  const int mlane = lane & 15;
  const int kg = (lane >> 4) & 1;
  const int kb = kg << 4;
  v8f au[4] = {}, ag[4] = {};
  const _Float16* arow = A + (size_t)(row0 + mlane) * K;
  for (int k0 = 0; k0 < K; k0 += 32) {
    if (k0 + 64 <= K) __builtin_prefetch((const void*)(arow + k0 + 32), 0, 0);
    const _Float16* ap = arow + k0 + (kg << 4);
    v16h af;
#pragma unroll
    for (int i = 0; i < 8; ++i) { af[i] = ap[i]; af[8 + i] = ap[8 + i]; }
#pragma unroll
    for (int tI = 0; tI < 4; ++tI) {
      const int nrow = col0 + (tI << 4) + mlane;
      const _Float16* bu = Bt + (size_t)nrow * K + k0 + kb;
      const _Float16* bg = Bt + (size_t)(Nh + nrow) * K + k0 + kb;
      v16h bfu, bfg;
#pragma unroll
      for (int i = 0; i < 16; ++i) { bfu[i] = bu[i]; bfg[i] = bg[i]; }
      au[tI] = wmma16(af, bfu, au[tI]);
      ag[tI] = wmma16(af, bfg, ag[tI]);
    }
  }
  const int mbase = (lane >> 4) << 3;
#pragma unroll
  for (int tI = 0; tI < 4; ++tI) {
#pragma unroll
    for (int r = 0; r < 8; ++r) {
      const int gm = row0 + mbase + r;
      const int gn = col0 + (tI << 4) + mlane;
      float u = au[tI][r] + b1[gn];
      float g = ag[tI][r] + b1[Nh + gn];
      float ge = 0.5f * g * (1.0f + erff(g * 0.70710678118f));  // exact GELU
      Y[(size_t)gm * Nh + gn] = (_Float16)(u * ge);
    }
  }
}

// ---------------- flash attention: one wave = 16 queries, 32 keys/step -----
// Q,K: f16 [B*Ntok, 512] row-major; Vt: f16 [(b*512+col)*Nk + tok].
__global__ __launch_bounds__(256) void attn_k(
    const _Float16* __restrict__ Q, const _Float16* __restrict__ Kmat,
    const _Float16* __restrict__ Vt, _Float16* __restrict__ O,
    int Nq, int Nk) {
  __shared__ _Float16 pbuf[8][16][40];   // per-wave P bounce (layout swap)
  const int wv = threadIdx.x >> 5, lane = threadIdx.x & 31;
  const int qblocks = Nq >> 7;
  const int qb = blockIdx.x % qblocks;
  const int bh = blockIdx.x / qblocks;
  const int hh = bh & (NHEAD - 1), b = bh >> 3;
  const int q0 = (qb << 7) + (wv << 4);
  const int mlane = lane & 15;
  const int kg = (lane >> 4) & 1;
  const int kb = kg << 4;
  // preload Q A-fragments for the two d-chunks (d=64 -> 2 x K=32)
  v16h qf[2];
  const _Float16* qrow =
      Q + (size_t)(b * Nq + q0 + mlane) * DMODEL + hh * HDIM;
#pragma unroll
  for (int c = 0; c < 2; ++c) {
    const _Float16* qp = qrow + c * 32 + (kg << 4);
#pragma unroll
    for (int i = 0; i < 8; ++i) { qf[c][i] = qp[i]; qf[c][8 + i] = qp[8 + i]; }
  }
  float mrow[8], lrow[8];
  v8f o[4] = {};
#pragma unroll
  for (int r = 0; r < 8; ++r) { mrow[r] = -1e30f; lrow[r] = 0.f; }

  for (int kv = 0; kv < Nk; kv += 32) {
    v8f s0 = {}, s1 = {};
#pragma unroll
    for (int c = 0; c < 2; ++c) {
      const _Float16* kp0 = Kmat + (size_t)(b * Nk + kv + mlane) * DMODEL +
                            hh * HDIM + c * 32 + kb;
      const _Float16* kp1 = kp0 + (size_t)16 * DMODEL;
      v16h bf0, bf1;
#pragma unroll
      for (int i = 0; i < 16; ++i) { bf0[i] = kp0[i]; bf1[i] = kp1[i]; }
      s0 = wmma16(qf[c], bf0, s0);
      s1 = wmma16(qf[c], bf1, s1);
    }
    // online softmax (rows live in VGPR index r, cols across 16-lane halves)
#pragma unroll
    for (int r = 0; r < 8; ++r) {
      float a0 = s0[r] * 0.125f;        // d^-0.5 = 1/8
      float a1 = s1[r] * 0.125f;
      float mx = fmaxf(a0, a1);
#pragma unroll
      for (int off = 1; off < 16; off <<= 1)
        mx = fmaxf(mx, __shfl_xor(mx, off, 32));
      float mn = fmaxf(mrow[r], mx);
      float al = __expf(mrow[r] - mn);
      float p0 = __expf(a0 - mn);
      float p1 = __expf(a1 - mn);
      float rs = p0 + p1;
#pragma unroll
      for (int off = 1; off < 16; off <<= 1) rs += __shfl_xor(rs, off, 32);
      lrow[r] = al * lrow[r] + rs;
      mrow[r] = mn;
#pragma unroll
      for (int tI = 0; tI < 4; ++tI) o[tI][r] *= al;
      const int mm = ((lane >> 4) << 3) + r;
      pbuf[wv][mm][mlane] = (_Float16)p0;
      pbuf[wv][mm][16 + mlane] = (_Float16)p1;
    }
    asm volatile("s_wait_dscnt 0" ::: "memory");   // wave-local LDS RAW
    v16h pf;
#pragma unroll
    for (int i = 0; i < 8; ++i) {
      pf[i]     = pbuf[wv][mlane][(kg << 4) + i];
      pf[8 + i] = pbuf[wv][mlane][(kg << 4) + 8 + i];
    }
#pragma unroll
    for (int tI = 0; tI < 4; ++tI) {
      const _Float16* vp =
          Vt + ((size_t)(b * DMODEL + hh * HDIM + (tI << 4) + mlane)) * Nk +
          kv + kb;
      v16h vf;
#pragma unroll
      for (int i = 0; i < 16; ++i) vf[i] = vp[i];
      o[tI] = wmma16(pf, vf, o[tI]);
    }
  }
  const int mbase = (lane >> 4) << 3;
#pragma unroll
  for (int tI = 0; tI < 4; ++tI) {
#pragma unroll
    for (int r = 0; r < 8; ++r) {
      const int gm = b * Nq + q0 + mbase + r;
      const int gn = hh * HDIM + (tI << 4) + mlane;
      O[(size_t)gm * DMODEL + gn] = (_Float16)(o[tI][r] / lrow[r]);
    }
  }
}

// ---------------------------------------------------------------------------
extern "C" void kernel_launch(void* const* d_in, const int* in_sizes, int n_in,
                              void* d_out, int out_size, void* d_ws,
                              size_t ws_size, hipStream_t stream) {
  (void)in_sizes; (void)n_in; (void)out_size; (void)ws_size;
  const float* x    = (const float*)d_in[0];
  const float* t    = (const float*)d_in[1];
  const float* cond = (const float*)d_in[2];
  const float* n1_w = (const float*)d_in[3];
  const float* n1_b = (const float*)d_in[4];
  const float* n2_w = (const float*)d_in[5];
  const float* n2_b = (const float*)d_in[6];
  const float* n4_w = (const float*)d_in[7];
  const float* n4_b = (const float*)d_in[8];
  const float* a1_q = (const float*)d_in[9];
  const float* a1_k = (const float*)d_in[10];
  const float* a1_v = (const float*)d_in[11];
  const float* a1_o = (const float*)d_in[12];
  const float* a1_ob= (const float*)d_in[13];
  const float* a2_q = (const float*)d_in[14];
  const float* a2_k = (const float*)d_in[15];
  const float* a2_v = (const float*)d_in[16];
  const float* a2_o = (const float*)d_in[17];
  const float* a2_ob= (const float*)d_in[18];
  const float* ffw1 = (const float*)d_in[19];
  const float* ffb1 = (const float*)d_in[20];
  const float* ffw2 = (const float*)d_in[21];
  const float* ffb2 = (const float*)d_in[22];
  float* out = (float*)d_out;

  // ---- workspace layout ----
  size_t off = 0;
  char* base = (char*)d_ws;
  auto alloc = [&](size_t bytes) -> char* {
    char* p = base + off;
    off += (bytes + 255) & ~(size_t)255;
    return p;
  };
  _Float16* h16   = (_Float16*)alloc((size_t)MROWS * DMODEL * 2);
  _Float16* q16   = (_Float16*)alloc((size_t)MROWS * DMODEL * 2);
  _Float16* k16   = (_Float16*)alloc((size_t)MROWS * DMODEL * 2);
  _Float16* vt16  = (_Float16*)alloc((size_t)NBATCH * DMODEL * NTOK * 2);
  _Float16* at16  = (_Float16*)alloc((size_t)MROWS * DMODEL * 2);
  _Float16* c16   = (_Float16*)alloc((size_t)NBATCH * NCTX * DMODEL * 2);
  float*    x1    = (float*)   alloc((size_t)MROWS * DMODEL * 4);
  float*    x2    = (float*)   alloc((size_t)MROWS * DMODEL * 4);
  _Float16* y16   = (_Float16*)alloc((size_t)MROWS * 2048 * 2);
  float*    ss    = (float*)   alloc((size_t)3 * NBATCH * 1024 * 4);
  _Float16* a1qT  = (_Float16*)alloc((size_t)DMODEL * DMODEL * 2);
  _Float16* a1kT  = (_Float16*)alloc((size_t)DMODEL * DMODEL * 2);
  _Float16* a1vT  = (_Float16*)alloc((size_t)DMODEL * DMODEL * 2);
  _Float16* a1oT  = (_Float16*)alloc((size_t)DMODEL * DMODEL * 2);
  _Float16* a2qT  = (_Float16*)alloc((size_t)DMODEL * DMODEL * 2);
  _Float16* a2kT  = (_Float16*)alloc((size_t)DMODEL * DMODEL * 2);
  _Float16* a2vT  = (_Float16*)alloc((size_t)DMODEL * DMODEL * 2);
  _Float16* a2oT  = (_Float16*)alloc((size_t)DMODEL * DMODEL * 2);
  _Float16* w1T   = (_Float16*)alloc((size_t)4096 * DMODEL * 2);   // [N=4096,K=512]
  _Float16* w2T   = (_Float16*)alloc((size_t)DMODEL * 2048 * 2);   // [N=512,K=2048]

  auto cvtT = [&](const float* w, _Float16* wt, int K, int N) {
    int n = N * K;
    cvtT_k<<<(n + 255) / 256, 256, 0, stream>>>(w, wt, K, N);
  };
  // weights -> fp16 transposed [N,K]
  cvtT(a1_q, a1qT, 512, 512);  cvtT(a1_k, a1kT, 512, 512);
  cvtT(a1_v, a1vT, 512, 512);  cvtT(a1_o, a1oT, 512, 512);
  cvtT(a2_q, a2qT, 512, 512);  cvtT(a2_k, a2kT, 512, 512);
  cvtT(a2_v, a2vT, 512, 512);  cvtT(a2_o, a2oT, 512, 512);
  cvtT(ffw1, w1T, 512, 4096);  cvtT(ffw2, w2T, 2048, 512);
  cvt_k<<<(NBATCH * NCTX * DMODEL + 255) / 256, 256, 0, stream>>>(
      cond, c16, NBATCH * NCTX * DMODEL);
  // adaln scale/shift embeddings for all 3 norms
  embed_k<<<(3 * NBATCH * 1024 + 255) / 256, 256, 0, stream>>>(
      t, n1_w, n1_b, n2_w, n2_b, n4_w, n4_b, ss);

  const int gM  = ((MROWS / 16) * (DMODEL / 64) + 7) / 8;   // 256 blocks
  const int gMc = (((NBATCH * NCTX) / 16) * (DMODEL / 64) + 7) / 8;
  const int gAttn = NBATCH * NHEAD * (NTOK / 128);          // 256 blocks

  // ---- stage 1: self-attention -------------------------------------------
  adaln_k<<<MROWS / 8, 256, 0, stream>>>(x, ss, 0, h16);
  gemm_k<0><<<gM, 256, 0, stream>>>(h16, a1qT, nullptr, nullptr, nullptr, q16,
                                    MROWS, DMODEL, DMODEL, 0);
  gemm_k<0><<<gM, 256, 0, stream>>>(h16, a1kT, nullptr, nullptr, nullptr, k16,
                                    MROWS, DMODEL, DMODEL, 0);
  gemm_k<1><<<gM, 256, 0, stream>>>(h16, a1vT, nullptr, nullptr, nullptr, vt16,
                                    MROWS, DMODEL, DMODEL, NTOK);
  attn_k<<<gAttn, 256, 0, stream>>>(q16, k16, vt16, at16, NTOK, NTOK);
  gemm_k<2><<<gM, 256, 0, stream>>>(at16, a1oT, a1_ob, x, x1, nullptr,
                                    MROWS, DMODEL, DMODEL, 0);

  // ---- stage 2: cross-attention ------------------------------------------
  adaln_k<<<MROWS / 8, 256, 0, stream>>>(x1, ss, 1, h16);
  gemm_k<0><<<gM, 256, 0, stream>>>(h16, a2qT, nullptr, nullptr, nullptr, q16,
                                    MROWS, DMODEL, DMODEL, 0);
  gemm_k<0><<<gMc, 256, 0, stream>>>(c16, a2kT, nullptr, nullptr, nullptr, k16,
                                     NBATCH * NCTX, DMODEL, DMODEL, 0);
  gemm_k<1><<<gMc, 256, 0, stream>>>(c16, a2vT, nullptr, nullptr, nullptr, vt16,
                                     NBATCH * NCTX, DMODEL, DMODEL, NCTX);
  attn_k<<<gAttn, 256, 0, stream>>>(q16, k16, vt16, at16, NTOK, NCTX);
  gemm_k<2><<<gM, 256, 0, stream>>>(at16, a2oT, a2_ob, x1, x2, nullptr,
                                    MROWS, DMODEL, DMODEL, 0);

  // ---- stage 3: GEGLU FF --------------------------------------------------
  adaln_k<<<MROWS / 8, 256, 0, stream>>>(x2, ss, 2, h16);
  const int gG = ((MROWS / 16) * (2048 / 64) + 7) / 8;      // 1024 blocks
  geglu_k<<<gG, 256, 0, stream>>>(h16, w1T, ffb1, y16, MROWS, DMODEL);
  gemm_k<2><<<gM, 256, 0, stream>>>(y16, w2T, ffb2, x2, out, nullptr,
                                    MROWS, DMODEL, 2048, 0);
}
